// MemoryWriteHead_84499186581790
// MI455X (gfx1250) — matrix operand order repaired and checked by reading, not compiled, not run on hardware
//
#include <hip/hip_runtime.h>
#include <hip/hip_bf16.h>
#include <math.h>

typedef __attribute__((ext_vector_type(16))) _Float16 v16h;
typedef __attribute__((ext_vector_type(8)))  float    v8f;

#define BB 256
#define NN 2048
#define MM 128
#define HH 512

#if defined(__has_builtin)
#  if __has_builtin(__builtin_amdgcn_global_load_async_to_lds_b128) && \
      __has_builtin(__builtin_amdgcn_s_wait_asynccnt)
#    define USE_ASYNC_LDS 1
#  endif
#  if __has_builtin(__builtin_amdgcn_rcpf)
#    define FAST_RCP(x) __builtin_amdgcn_rcpf(x)   // single v_rcp_f32
#  endif
#endif
#ifndef FAST_RCP
#  define FAST_RCP(x) (1.f / (x))
#endif

#ifdef USE_ASYNC_LDS
// Builtin signature (from hipcc diagnostics):
//   param0: int4 (vector_size 16) pointer, global  (__device__ / addrspace(1))
//   param1: int4 (vector_size 16) pointer, shared  (__shared__ / addrspace(3))
//   param2/3: imm offset, imm cpol
typedef int v4i_vs __attribute__((vector_size(16)));
typedef v4i_vs __attribute__((address_space(1))) * async_gptr;
typedef v4i_vs __attribute__((address_space(3))) * async_lptr;
#endif

__device__ __forceinline__ float fast_sigmoid(float x) {
    return FAST_RCP(1.f + __expf(-x));   // v_exp + v_add + v_rcp
}

// ---------------------------------------------------------------------------
// Kernel 1: projections  P = h @ W^T + b  for W in {Wk, We, Wa}
// One wave (32 lanes) per 16x16 output tile, v_wmma_f32_16x16x32_f16, K=512.
// Column tiles 0..7 -> write_key (ws), 8..15 -> sigmoid -> erase_vec (out),
// 16..23 -> add_vec (out).
// ---------------------------------------------------------------------------
__global__ __launch_bounds__(32) void proj_wmma_kernel(
    const float* __restrict__ h,
    const float* __restrict__ Wk, const float* __restrict__ bk,
    const float* __restrict__ We, const float* __restrict__ be,
    const float* __restrict__ Wa, const float* __restrict__ ba,
    float* __restrict__ key_ws,
    float* __restrict__ out_erase,
    float* __restrict__ out_add)
{
    const int tiles_n = (3 * MM) / 16;          // 24
    int tile = blockIdx.x;                      // 0..383
    int tm = tile / tiles_n;                    // batch-row tile 0..15
    int tn = tile % tiles_n;                    // column tile 0..23
    int lane = threadIdx.x;                     // 0..31, wave32
    int group = tn >> 3;                        // 0=Wk 1=We 2=Wa
    const float* W    = (group == 0) ? Wk : (group == 1) ? We : Wa;
    const float* bias = (group == 0) ? bk : (group == 1) ? be : ba;
    int ncol = ((tn & 7) << 4) + (lane & 15);   // column within M
    int mrow = (tm << 4) + (lane & 15);         // batch row for A fragment
    int hi   = (lane >> 4) & 1;                 // lane half

    v8f c = {};
#pragma unroll 2
    for (int k0 = 0; k0 < HH; k0 += 32) {
        v16h a, bfrag;
        // A (16x32 f16): lane row = mrow; K = (e>>3)*16 + hi*8 + (e&7)
        const float* hrow = h + mrow * HH + k0 + hi * 8;
#pragma unroll
        for (int e = 0; e < 16; ++e) {
            int K = ((e >> 3) << 4) + (e & 7);
            a[e] = (_Float16)hrow[K];
        }
        // B (32x16 f16): lane col = ncol; K = (e>>3)*16 + hi*8 + (e&7)  (B[k][n] = W[n][k])
        const float* wrow = W + ncol * HH + k0 + hi * 8;
#pragma unroll
        for (int e = 0; e < 16; ++e) bfrag[e] = (_Float16)wrow[((e >> 3) << 4) + (e & 7)];

        c = __builtin_amdgcn_wmma_f32_16x16x32_f16(
                false, a, false, bfrag, (short)0, c, false, false);
    }

    float bv = bias[ncol];
#pragma unroll
    for (int r = 0; r < 8; ++r) {
        int row = (tm << 4) + r + (hi << 3);    // D: M = r + 8*hi, N = lane&15
        float v = c[r] + bv;
        if (group == 0)      key_ws  [row * MM + ncol] = v;
        else if (group == 1) out_erase[row * MM + ncol] = fast_sigmoid(v);
        else                 out_add [row * MM + ncol] = v;
    }
}

// ---------------------------------------------------------------------------
// Kernel 2: scalar projections (strength / write_gate / alloc_gate)
// One wave per batch row; wave32 shuffle reductions over H=512.
// ---------------------------------------------------------------------------
__global__ __launch_bounds__(32) void scalars_kernel(
    const float* __restrict__ h,
    const float* __restrict__ Ws,  const float* __restrict__ bs,
    const float* __restrict__ Wg,  const float* __restrict__ bg,
    const float* __restrict__ Wag, const float* __restrict__ bag,
    float* __restrict__ strength, float* __restrict__ wgate,
    float* __restrict__ agate)
{
    int b = blockIdx.x, lane = threadIdx.x;
    const float* hb = h + b * HH;
    float s0 = 0.f, s1 = 0.f, s2 = 0.f;
    for (int k = lane; k < HH; k += 32) {
        float hv = hb[k];
        s0 += hv * Ws[k];
        s1 += hv * Wg[k];
        s2 += hv * Wag[k];
    }
#pragma unroll
    for (int off = 16; off; off >>= 1) {
        s0 += __shfl_xor(s0, off, 32);
        s1 += __shfl_xor(s1, off, 32);
        s2 += __shfl_xor(s2, off, 32);
    }
    if (lane == 0) {
        float x = s0 + bs[0];
        strength[b] = (x > 20.f) ? x : log1pf(__expf(x));     // softplus
        wgate[b] = fast_sigmoid(s1 + bg[0]);
        agate[b] = fast_sigmoid(s2 + bag[0]);
    }
}

// ---------------------------------------------------------------------------
// Kernel 3: cosine content addressing + softmax. One block per batch row.
// The single 256MB streaming pass -> bandwidth-bound floor (~11 us @23.3TB/s).
// CDNA5 path: async global->LDS DMA (ASYNCcnt), double-buffered 16-row (8KB)
// stages; each wave issues its own copies, waits s_wait_asynccnt, then a
// workgroup barrier makes the stage visible to all 8 waves. Fallback path:
// direct global_load_b128 per lane.
// ---------------------------------------------------------------------------
__global__ __launch_bounds__(256) void content_kernel(
    const float* __restrict__ memory,
    const float* __restrict__ key_ws,
    const float* __restrict__ strength,
    float* __restrict__ content)
{
    __shared__ __align__(16) float s_key[MM];
    __shared__ float s_sim[NN];
    __shared__ float s_red[256];
    __shared__ float s_knorm;
#ifdef USE_ASYNC_LDS
    __shared__ __align__(16) float s_stage[2][16 * MM];   // 2 x 8KB
#endif

    int b = blockIdx.x, t = threadIdx.x;
    int lane = t & 31, wave = t >> 5;           // 8 waves

    if (t < MM) s_key[t] = key_ws[b * MM + t];
    __syncthreads();

    if (wave == 0) {
        float ss = 0.f;
        for (int i = lane; i < MM; i += 32) { float v = s_key[i]; ss += v * v; }
#pragma unroll
        for (int off = 16; off; off >>= 1) ss += __shfl_xor(ss, off, 32);
        if (lane == 0) s_knorm = sqrtf(ss);
    }
    __syncthreads();

    float knorm = s_knorm;
    float beta  = strength[b];
    const float* mb = memory + (size_t)b * NN * MM;
    float4 kf = ((const float4*)s_key)[lane];   // 4 key elems per lane

#ifdef USE_ASYNC_LDS
    const int SR = 16;                          // rows per stage
    const int NSTAGE = NN / SR;                 // 128 stages
    // prologue: issue stage 0 (each thread copies 2 x 16B; 2 async ops/wave)
    {
#pragma unroll
        for (int q = 0; q < 2; ++q) {
            int idx = (t + q * 256) * 4;        // float index within stage
            __builtin_amdgcn_global_load_async_to_lds_b128(
                (async_gptr)(float*)(mb + idx),
                (async_lptr)(&s_stage[0][idx]),
                0, 0);
        }
    }
    for (int st = 0; st < NSTAGE; ++st) {
        if (st + 1 < NSTAGE) {
            const float* g = mb + (size_t)(st + 1) * SR * MM;
            float* l = s_stage[(st + 1) & 1];
#pragma unroll
            for (int q = 0; q < 2; ++q) {
                int idx = (t + q * 256) * 4;
                __builtin_amdgcn_global_load_async_to_lds_b128(
                    (async_gptr)(float*)(g + idx),
                    (async_lptr)(&l[idx]),
                    0, 0);
            }
            __builtin_amdgcn_s_wait_asynccnt(2);   // stage st landed (this wave)
        } else {
            __builtin_amdgcn_s_wait_asynccnt(0);
        }
        __syncthreads();                            // cross-wave visibility
        const float* lbuf = s_stage[st & 1];
        for (int r = wave; r < SR; r += 8) {
            float4 mv = ((const float4*)(lbuf + r * MM))[lane];
            float dot = mv.x * kf.x + mv.y * kf.y + mv.z * kf.z + mv.w * kf.w;
            float sq  = mv.x * mv.x + mv.y * mv.y + mv.z * mv.z + mv.w * mv.w;
#pragma unroll
            for (int off = 16; off; off >>= 1) {
                dot += __shfl_xor(dot, off, 32);
                sq  += __shfl_xor(sq,  off, 32);
            }
            if (lane == 0)
                s_sim[st * SR + r] = beta * (dot / (knorm * sqrtf(sq) + 1e-8f));
        }
        __syncthreads();                            // buffer reuse fence
    }
#else
    for (int n = wave; n < NN; n += 8) {
        const float4* row = (const float4*)(mb + (size_t)n * MM);
        __builtin_prefetch(mb + (size_t)(n + 8) * MM + lane * 4, 0, 0);
        float4 mv = row[lane];
        float dot = mv.x * kf.x + mv.y * kf.y + mv.z * kf.z + mv.w * kf.w;
        float sq  = mv.x * mv.x + mv.y * mv.y + mv.z * mv.z + mv.w * mv.w;
#pragma unroll
        for (int off = 16; off; off >>= 1) {
            dot += __shfl_xor(dot, off, 32);
            sq  += __shfl_xor(sq,  off, 32);
        }
        if (lane == 0)
            s_sim[n] = beta * (dot / (knorm * sqrtf(sq) + 1e-8f));
    }
    __syncthreads();
#endif

    // softmax over s_sim[0..N)
    float mx = -3.402823466e38f;
    for (int i = t; i < NN; i += 256) mx = fmaxf(mx, s_sim[i]);
    s_red[t] = mx; __syncthreads();
    for (int off = 128; off; off >>= 1) {
        if (t < off) s_red[t] = fmaxf(s_red[t], s_red[t + off]);
        __syncthreads();
    }
    mx = s_red[0]; __syncthreads();

    float sum = 0.f;
    for (int i = t; i < NN; i += 256) {
        float e = __expf(s_sim[i] - mx);
        s_sim[i] = e; sum += e;
    }
    s_red[t] = sum; __syncthreads();
    for (int off = 128; off; off >>= 1) {
        if (t < off) s_red[t] += s_red[t + off];
        __syncthreads();
    }
    float inv = FAST_RCP(s_red[0]);
    for (int i = t; i < NN; i += 256)
        content[(size_t)b * NN + i] = s_sim[i] * inv;
}

// ---------------------------------------------------------------------------
// Kernel 4: DNC allocation weights + final gate combine. One block (1024 thr,
// 32 waves) per batch row. LDS bitonic sort (stable via index tie-break),
// Hillis-Steele inclusive cumprod, scatter, fused write_weights.
// ---------------------------------------------------------------------------
__global__ __launch_bounds__(1024) void alloc_kernel(
    const float* __restrict__ prev_usage,
    const float* __restrict__ content,
    const float* __restrict__ wgate,
    const float* __restrict__ agate,
    float* __restrict__ out_ww,
    float* __restrict__ out_alloc)
{
    __shared__ float s_u[NN];
    __shared__ int   s_idx[NN];
    __shared__ float s_pa[NN];
    __shared__ float s_pb[NN];

    int b = blockIdx.x, t = threadIdx.x;
    const float* ub = prev_usage + (size_t)b * NN;

    for (int i = t; i < NN; i += 1024) { s_u[i] = ub[i]; s_idx[i] = i; }
    __syncthreads();

    // bitonic sort ascending by (usage, index)
    for (int k = 2; k <= NN; k <<= 1) {
        for (int j = k >> 1; j > 0; j >>= 1) {
            for (int e = t; e < NN; e += 1024) {
                int l = e ^ j;
                if (l > e) {
                    bool asc = ((e & k) == 0);
                    float ue = s_u[e], ul = s_u[l];
                    int   ie = s_idx[e], il = s_idx[l];
                    bool gt = (ue > ul) || (ue == ul && ie > il);
                    if (gt == asc) {
                        s_u[e] = ul;  s_u[l] = ue;
                        s_idx[e] = il; s_idx[l] = ie;
                    }
                }
            }
            __syncthreads();
        }
    }

    // inclusive cumprod of sorted usage (Hillis-Steele, ping-pong)
    for (int i = t; i < NN; i += 1024) s_pa[i] = s_u[i];
    __syncthreads();
    float* pa = s_pa; float* pb = s_pb;
    for (int off = 1; off < NN; off <<= 1) {
        for (int i = t; i < NN; i += 1024)
            pb[i] = (i >= off) ? pa[i] * pa[i - off] : pa[i];
        __syncthreads();
        float* tmp = pa; pa = pb; pb = tmp;
    }

    float wg = wgate[b], ag = agate[b];
    const float* cb = content + (size_t)b * NN;
    for (int i = t; i < NN; i += 1024) {
        float excl  = (i == 0) ? 1.f : pa[i - 1];       // exclusive cumprod
        float alloc = (1.f - s_u[i]) * excl;
        int n = s_idx[i];
        out_alloc[(size_t)b * NN + n] = alloc;
        out_ww  [(size_t)b * NN + n] = wg * (ag * alloc + (1.f - ag) * cb[n]);
    }
}

// ---------------------------------------------------------------------------
extern "C" void kernel_launch(void* const* d_in, const int* in_sizes, int n_in,
                              void* d_out, int out_size, void* d_ws, size_t ws_size,
                              hipStream_t stream) {
    (void)in_sizes; (void)n_in; (void)out_size; (void)ws_size;
    const float* h          = (const float*)d_in[0];
    const float* memory     = (const float*)d_in[1];
    const float* prev_usage = (const float*)d_in[2];
    const float* Wk  = (const float*)d_in[3];  const float* bk  = (const float*)d_in[4];
    const float* Ws  = (const float*)d_in[5];  const float* bs  = (const float*)d_in[6];
    const float* We  = (const float*)d_in[7];  const float* be  = (const float*)d_in[8];
    const float* Wa  = (const float*)d_in[9];  const float* ba  = (const float*)d_in[10];
    const float* Wg  = (const float*)d_in[11]; const float* bg  = (const float*)d_in[12];
    const float* Wag = (const float*)d_in[13]; const float* bag = (const float*)d_in[14];

    float* out = (float*)d_out;
    float* out_ww    = out;                         // (B,N)
    float* out_erase = out + BB * NN;               // (B,M)
    float* out_add   = out + BB * NN + BB * MM;     // (B,M)
    float* out_alloc = out + BB * NN + 2 * BB * MM; // (B,N)

    float* ws       = (float*)d_ws;
    float* key_ws   = ws;                           // B*M
    float* strength = ws + BB * MM;                 // B
    float* wgate    = strength + BB;                // B
    float* agate    = wgate + BB;                   // B
    float* content  = agate + BB;                   // B*N

    proj_wmma_kernel<<<(BB / 16) * ((3 * MM) / 16), 32, 0, stream>>>(
        h, Wk, bk, We, be, Wa, ba, key_ws, out_erase, out_add);
    scalars_kernel<<<BB, 32, 0, stream>>>(
        h, Ws, bs, Wg, bg, Wag, bag, strength, wgate, agate);
    content_kernel<<<BB, 256, 0, stream>>>(memory, key_ws, strength, content);
    alloc_kernel<<<BB, 1024, 0, stream>>>(
        prev_usage, content, wgate, agate, out_ww, out_alloc);
}
